// AdaptiveSamplingMixing_77249281786079
// MI455X (gfx1250) — matrix operand order repaired and checked by reading, not compiled, not run on hardware
//
#include <hip/hip_runtime.h>

// ---------------------------------------------------------------------------
// AdaptiveSamplingMixing for MI455X (gfx1250), wave32 + WMMA bf16 path.
// Round 3: weights converted to bf16 *pre-transposed* (N-major) so GEMM
// B-fragments are two global_load_b128 per lane straight from L2 — no
// in-loop LDS transpose scatter, barrier-free params GEMM inner loop.
// ---------------------------------------------------------------------------

#define DB 2
#define DN 300
#define BN 600          // B*N
#define CQ 256
#define GG 4
#define PIN 32
#define POUT 128
#define CGc 64          // channels per group
#define TOTAL 32768     // G*(Cg*Cg + P_OUT*P_IN)
#define PERG 8192       // Cg*Cg + P_OUT*P_IN

typedef __attribute__((ext_vector_type(16))) __bf16          bf16x16;
typedef __attribute__((ext_vector_type(8)))  float           f32x8;

struct U4x2 { uint4 lo, hi; };   // 32 bytes == one bf16x16 fragment

// ---- fp32 -> bf16 round-to-nearest-even -----------------------------------
static __device__ inline unsigned short f2bf(float f) {
  unsigned u = __builtin_bit_cast(unsigned, f);
  unsigned r = u + 0x7FFFu + ((u >> 16) & 1u);
  return (unsigned short)(r >> 16);
}

// ---- WMMA fragment builders (CDNA5 ISA 7.12.2 layouts, vectorized) ---------
// A: 16x32 window of row-major bf16 (lda elems). lane<16 -> K {0..7,16..23}.
static __device__ inline bf16x16 load_a_frag(const unsigned short* src, int lda, int lane) {
  int m   = lane & 15;
  int klo = (lane < 16) ? 0 : 8;
  const unsigned short* row = src + m * lda + klo;
  U4x2 t;
  t.lo = *(const uint4*)(row);        // K = klo .. klo+7
  t.hi = *(const uint4*)(row + 16);   // K = klo+16 .. klo+23
  return __builtin_bit_cast(bf16x16, t);
}

// B: 32x16 (KxN) window of a TRANSPOSED matrix stored [n][k] (ldk elems).
// lane<16 -> col n=lane, K {0..7,16..23}; lanes 16..31 -> K {8..15,24..31}.
static __device__ inline bf16x16 load_bT_frag(const unsigned short* src, int ldk, int lane) {
  int n  = lane & 15;
  int kb = (lane < 16) ? 0 : 8;
  const unsigned short* row = src + n * ldk + kb;
  U4x2 t;
  t.lo = *(const uint4*)(row);        // K = kb .. kb+7
  t.hi = *(const uint4*)(row + 16);   // K = kb+16 .. kb+23
  return __builtin_bit_cast(bf16x16, t);
}

static __device__ inline f32x8 wmma_bf16(bf16x16 a, bf16x16 b, f32x8 c) {
  return __builtin_amdgcn_wmma_f32_16x16x32_bf16(false, a, false, b, (short)0, c, false, false);
}

static __device__ inline float wave_sum(float v) {
#pragma unroll
  for (int o = 16; o > 0; o >>= 1) v += __shfl_xor(v, o, 32);
  return v;
}

// ---------------------------------------------------------------------------
// K1a: fp32 -> bf16 conversion (row-major, 4 elems/thread)
// ---------------------------------------------------------------------------
__global__ __launch_bounds__(256) void cvt_bf16(const float* __restrict__ src,
                                                unsigned short* __restrict__ dst, int n4) {
  for (int i = blockIdx.x * blockDim.x + threadIdx.x; i < n4; i += gridDim.x * blockDim.x) {
    float4 v = ((const float4*)src)[i];
    ushort4 o;
    o.x = f2bf(v.x); o.y = f2bf(v.y); o.z = f2bf(v.z); o.w = f2bf(v.w);
    ((ushort4*)dst)[i] = o;
  }
}

// ---------------------------------------------------------------------------
// K1b: transpose + convert: W(K,N) fp32 row-major -> Wt(N,K) bf16 row-major.
// LDS-tiled 64x64: coalesced float4 reads, coalesced uint4 writes.
// K, N multiples of 64. grid = (N/64, K/64)
// ---------------------------------------------------------------------------
__global__ __launch_bounds__(256) void transpose_cvt(const float* __restrict__ W,
                                                     unsigned short* __restrict__ Wt,
                                                     int K, int N) {
  int n0 = blockIdx.x * 64;
  int k0 = blockIdx.y * 64;
  int tid = threadIdx.x;
  __shared__ __align__(16) unsigned short T[64 * 72];   // [k][n], stride 72
#pragma unroll
  for (int j = 0; j < 4; ++j) {
    int u  = tid + j * 256;              // u in [0,1024): float4 groups
    int rr = u >> 4, c4 = (u & 15) * 4;  // rr = k row, c4 = n col group
    float4 v = *(const float4*)&W[(size_t)(k0 + rr) * N + n0 + c4];
    ushort4 o;
    o.x = f2bf(v.x); o.y = f2bf(v.y); o.z = f2bf(v.z); o.w = f2bf(v.w);
    *(ushort4*)&T[rr * 72 + c4] = o;
  }
  __syncthreads();
#pragma unroll
  for (int j = 0; j < 2; ++j) {
    int u  = tid + j * 256;              // u in [0,512): uint4 groups of Wt
    int nn = u >> 3, k8 = (u & 7) * 8;   // nn = n row, k8 = k col group
    unsigned short tmp[8];
#pragma unroll
    for (int i = 0; i < 8; ++i) tmp[i] = T[(k8 + i) * 72 + nn];
    *(uint4*)&Wt[(size_t)(n0 + nn) * K + k0 + k8] = *(const uint4*)tmp;
  }
}

// ---------------------------------------------------------------------------
// K2: offsets + sampling coords + level softmax weights.
// ---------------------------------------------------------------------------
__global__ __launch_bounds__(128) void offset_kernel(const float* __restrict__ qf,
                                                     const float* __restrict__ roi,
                                                     const float* __restrict__ w_off,
                                                     const float* __restrict__ b_off,
                                                     float* __restrict__ coords) {
  int bn = blockIdx.x;
  int t  = threadIdx.x;      // (g*32+p) in [0,128)
  __shared__ float sq[CQ];
  sq[t]       = qf[bn * CQ + t];
  sq[t + 128] = qf[bn * CQ + t + 128];
  __syncthreads();

  float o0 = b_off[t * 3 + 0], o1 = b_off[t * 3 + 1], o2 = b_off[t * 3 + 2];
  for (int k = 0; k < CQ; ++k) {
    float q = sq[k];
    const float* wr = w_off + (size_t)k * 384 + t * 3;
    o0 = fmaf(q, wr[0], o0);
    o1 = fmaf(q, wr[1], o1);
    o2 = fmaf(q, wr[2], o2);
  }
  float cx = roi[bn * 4 + 0], cy = roi[bn * 4 + 1];
  float z  = roi[bn * 4 + 2], r  = roi[bn * 4 + 3];
  float rw = exp2f(z - 0.5f * r);
  float rh = exp2f(z + 0.5f * r);
  float sx = cx + o0 * rw;
  float sy = cy + o1 * rh;
  float lvl = z + o2;

  float s[4], mx = -1e30f;
#pragma unroll
  for (int l = 0; l < 4; ++l) {
    float d = lvl - 3.0f - (float)l;
    s[l] = -d * d * 0.5f;          // TAU = 2.0
    mx = fmaxf(mx, s[l]);
  }
  float e[4], den = 0.f;
#pragma unroll
  for (int l = 0; l < 4; ++l) { e[l] = expf(s[l] - mx); den += e[l]; }
  float inv = 1.0f / den;

  float* rec = coords + ((size_t)bn * 128 + t) * 6;
  rec[0] = sx; rec[1] = sy;
#pragma unroll
  for (int l = 0; l < 4; ++l) rec[2 + l] = e[l] * inv;
}

// ---------------------------------------------------------------------------
// K3: params GEMM  C(600x32768) = qf_bf(600x256) @ pg_w(256x32768) + pg_b
// B pre-transposed (WtP[n][k], K=256). A panel staged once, 8 A-frags in
// registers, barrier-free inner loop: 16 WMMA per wave per block.
// ---------------------------------------------------------------------------
__global__ __launch_bounds__(256) void param_gemm(const unsigned short* __restrict__ A,
                                                  const unsigned short* __restrict__ WtP,
                                                  const float* __restrict__ bias,
                                                  unsigned short* __restrict__ Cbf) {
  int n0  = blockIdx.x * 256;
  int m0  = blockIdx.y * 16;
  int tid = threadIdx.x, lane = tid & 31, wv = tid >> 5;
  __shared__ __align__(16) unsigned short As[16 * 256];   // 8 KB; reused as C staging

  // stage full 16x256 A panel (512 x uint4)
#pragma unroll
  for (int j = 0; j < 2; ++j) {
    int u  = tid + j * 256;
    int rr = u >> 5, c8 = (u & 31) * 8;
    int m  = m0 + rr;
    uint4 g = (m < BN) ? *(const uint4*)&A[(size_t)m * CQ + c8] : uint4{0, 0, 0, 0};
    *(uint4*)&As[rr * 256 + c8] = g;
  }
  __syncthreads();

  // preload all 8 A-fragments
  bf16x16 af[8];
#pragma unroll
  for (int t = 0; t < 8; ++t) af[t] = load_a_frag(As + t * 32, 256, lane);

  const unsigned short* b0p = WtP + (size_t)(n0 + wv * 32) * CQ;
  const unsigned short* b1p = b0p + (size_t)16 * CQ;
  f32x8 acc0 = {}, acc1 = {};
#pragma unroll
  for (int t = 0; t < 8; ++t) {
    bf16x16 b0 = load_bT_frag(b0p + t * 32, CQ, lane);
    bf16x16 b1 = load_bT_frag(b1p + t * 32, CQ, lane);
    acc0 = wmma_bf16(af[t], b0, acc0);
    acc1 = wmma_bf16(af[t], b1, acc1);
  }

  // epilogue: bias add, stage bf16 tile in LDS (reuse As), coalesced stores
  float bv0 = bias[n0 + wv * 32 + (lane & 15)];
  float bv1 = bias[n0 + wv * 32 + 16 + (lane & 15)];
  __syncthreads();   // all waves done reading As
#pragma unroll
  for (int rI = 0; rI < 8; ++rI) {
    int ml = rI + 8 * (lane >> 4);
    As[ml * 256 + wv * 32 + (lane & 15)]      = f2bf(acc0[rI] + bv0);
    As[ml * 256 + wv * 32 + 16 + (lane & 15)] = f2bf(acc1[rI] + bv1);
  }
  __syncthreads();
#pragma unroll
  for (int j = 0; j < 2; ++j) {
    int u  = tid + j * 256;
    int rr = u >> 5, c8 = (u & 31) * 8;
    int m  = m0 + rr;
    if (m < BN)
      *(uint4*)&Cbf[(size_t)m * TOTAL + n0 + c8] = *(const uint4*)&As[rr * 256 + c8];
  }
}

// ---------------------------------------------------------------------------
// K4: multi-level bilinear sampling (zero-pad OOB, per-corner validity).
// ---------------------------------------------------------------------------
__global__ __launch_bounds__(256) void sample_kernel(const float* __restrict__ x0,
                                                     const float* __restrict__ x1,
                                                     const float* __restrict__ x2,
                                                     const float* __restrict__ x3,
                                                     const float* __restrict__ coords,
                                                     unsigned short* __restrict__ sampled) {
  int bn = blockIdx.x, g = blockIdx.y;
  int b  = bn / DN;
  const float* xs[4] = {x0, x1, x2, x3};
  const int Hs[4] = {200, 100, 50, 25};
  const int Ws[4] = {336, 168, 84, 42};
  const float inv_s[4] = {0.25f, 0.125f, 0.0625f, 0.03125f};

  int c  = threadIdx.x & 63;
  int p0 = threadIdx.x >> 6;
  for (int p = p0; p < PIN; p += 4) {
    const float* rec = coords + ((size_t)bn * 128 + g * PIN + p) * 6;
    float sx = rec[0], sy = rec[1];
    float acc = 0.f;
#pragma unroll
    for (int l = 0; l < 4; ++l) {
      float w = rec[2 + l];
      int H = Hs[l], W = Ws[l];
      float gx = sx * inv_s[l] - 0.5f;
      float gy = sy * inv_s[l] - 0.5f;
      float xf = floorf(gx), yf = floorf(gy);
      int xi = (int)xf, yi = (int)yf;
      float fx = gx - xf, fy = gy - yf;
      const float* base = xs[l] + ((size_t)(b * CQ + g * CGc + c)) * H * W;
      bool vx0 = (xi >= 0) & (xi < W), vx1 = (xi + 1 >= 0) & (xi + 1 < W);
      bool vy0 = (yi >= 0) & (yi < H), vy1 = (yi + 1 >= 0) & (yi + 1 < H);
      float v00 = (vx0 & vy0) ? base[yi * W + xi]           : 0.f;
      float v10 = (vx1 & vy0) ? base[yi * W + xi + 1]       : 0.f;
      float v01 = (vx0 & vy1) ? base[(yi + 1) * W + xi]     : 0.f;
      float v11 = (vx1 & vy1) ? base[(yi + 1) * W + xi + 1] : 0.f;
      float bil = v00 * (1.f - fx) * (1.f - fy) + v10 * fx * (1.f - fy)
                + v01 * (1.f - fx) * fy         + v11 * fx * fy;
      acc = fmaf(w, bil, acc);
    }
    sampled[(((size_t)bn * GG + g) * PIN + p) * CGc + c] = f2bf(acc);
  }
}

// ---------------------------------------------------------------------------
// K5: per-(b,n,g) mixing (WMMA x2 + LN/ReLU), M and h1 transposed in LDS.
// ---------------------------------------------------------------------------
__global__ __launch_bounds__(256) void mix_kernel(const unsigned short* __restrict__ sampled,
                                                  const unsigned short* __restrict__ params,
                                                  unsigned short* __restrict__ hflat) {
  int bn = blockIdx.x, g = blockIdx.y;
  int tid = threadIdx.x, lane = tid & 31, wv = tid >> 5;

  __shared__ __align__(16) unsigned short sA[32 * 64];    // sampled 32x64 (p x c)
  __shared__ __align__(16) unsigned short Mt[64 * 64];    // M transposed (d x c)
  __shared__ __align__(16) unsigned short sS[128 * 32];   // S 128x32 (o x p)
  __shared__ __align__(16) unsigned short sHt[64 * 32];   // h1 transposed (c x p)
  __shared__ __align__(16) unsigned short sOut[128 * 64]; // h2 staging (o x c)
  __shared__ float sred[16];
  __shared__ float sstat[2];

  const unsigned short* smp = sampled + ((size_t)bn * GG + g) * PIN * CGc;
  const unsigned short* prm = params + (size_t)bn * TOTAL + (size_t)g * PERG;

  ((uint4*)sA)[tid] = ((const uint4*)smp)[tid];
  ((uint4*)sS)[tid]       = ((const uint4*)(prm + 4096))[tid];
  ((uint4*)sS)[tid + 256] = ((const uint4*)(prm + 4096))[tid + 256];
#pragma unroll
  for (int j = 0; j < 2; ++j) {
    int u  = tid + j * 256;              // uint4 index into M
    int cc = u >> 3, d8 = (u & 7) * 8;
    uint4 gv = *(const uint4*)&prm[cc * 64 + d8];
    const unsigned short* gs = (const unsigned short*)&gv;
#pragma unroll
    for (int i = 0; i < 8; ++i) Mt[(d8 + i) * 64 + cc] = gs[i];
  }
  __syncthreads();

  // ---- h1 = sampled @ M : 2x4 tiles, wave -> one tile, K=64
  int pt0 = (wv >> 2) * 16;
  int d0  = (wv & 3) * 16;
  f32x8 acc1 = {};
#pragma unroll
  for (int k0 = 0; k0 < 64; k0 += 32) {
    bf16x16 a = load_a_frag(sA + pt0 * 64 + k0, 64, lane);
    bf16x16 b = load_bT_frag(Mt + d0 * 64 + k0, 64, lane);
    acc1 = wmma_bf16(a, b, acc1);
  }

  // ---- LN over all 2048 elements of h1
  float ls = 0.f, lq = 0.f;
#pragma unroll
  for (int rI = 0; rI < 8; ++rI) { float v = acc1[rI]; ls += v; lq += v * v; }
  ls = wave_sum(ls); lq = wave_sum(lq);
  if (lane == 0) { sred[wv] = ls; sred[8 + wv] = lq; }
  __syncthreads();
  if (tid == 0) {
    float s = 0.f, q = 0.f;
    for (int i = 0; i < 8; ++i) { s += sred[i]; q += sred[8 + i]; }
    float mean = s * (1.f / 2048.f);
    float var  = q * (1.f / 2048.f) - mean * mean;
    sstat[0] = mean; sstat[1] = rsqrtf(var + 1e-5f);
  }
  __syncthreads();
  float mean1 = sstat[0], rstd1 = sstat[1];
#pragma unroll
  for (int rI = 0; rI < 8; ++rI) {
    int p = pt0 + rI + 8 * (lane >> 4);
    int c = d0 + (lane & 15);
    float v = (acc1[rI] - mean1) * rstd1;
    sHt[c * 32 + p] = f2bf(fmaxf(v, 0.f));     // transposed store
  }
  __syncthreads();

  // ---- h2 = S @ h1 : 8x4 tiles, wave -> o-tile row, K=32
  int o0 = wv * 16;
  bf16x16 a2 = load_a_frag(sS + o0 * 32, 32, lane);
  f32x8 acc2[4];
#pragma unroll
  for (int j = 0; j < 4; ++j) {
    bf16x16 b2 = load_bT_frag(sHt + (j * 16) * 32, 32, lane);
    f32x8 z = {};
    acc2[j] = wmma_bf16(a2, b2, z);
  }

  // ---- LN over all 8192 elements of h2
  ls = 0.f; lq = 0.f;
#pragma unroll
  for (int j = 0; j < 4; ++j)
#pragma unroll
    for (int rI = 0; rI < 8; ++rI) { float v = acc2[j][rI]; ls += v; lq += v * v; }
  ls = wave_sum(ls); lq = wave_sum(lq);
  if (lane == 0) { sred[wv] = ls; sred[8 + wv] = lq; }
  __syncthreads();
  if (tid == 0) {
    float s = 0.f, q = 0.f;
    for (int i = 0; i < 8; ++i) { s += sred[i]; q += sred[8 + i]; }
    float mean = s * (1.f / 8192.f);
    float var  = q * (1.f / 8192.f) - mean * mean;
    sstat[0] = mean; sstat[1] = rsqrtf(var + 1e-5f);
  }
  __syncthreads();
  float mean2 = sstat[0], rstd2 = sstat[1];
#pragma unroll
  for (int j = 0; j < 4; ++j)
#pragma unroll
    for (int rI = 0; rI < 8; ++rI) {
      int o = o0 + rI + 8 * (lane >> 4);
      int c = j * 16 + (lane & 15);
      float v = (acc2[j][rI] - mean2) * rstd2;
      sOut[o * CGc + c] = f2bf(fmaxf(v, 0.f));
    }
  __syncthreads();
  unsigned short* outp = hflat + (size_t)bn * TOTAL + (size_t)g * PERG;
#pragma unroll
  for (int j = 0; j < 4; ++j) {
    int u = tid + j * 256;
    ((uint4*)outp)[u] = ((const uint4*)sOut)[u];
  }
}

// ---------------------------------------------------------------------------
// K6: output GEMM  qpre(600x256) = hflat(600x32768) @ op_w(32768x256)
// B pre-transposed (WtO[n][k], K=32768) read directly from global/L2.
// A staged in 256-deep K chunks; + op_b + residual, coalesced fp32 out.
// ---------------------------------------------------------------------------
__global__ __launch_bounds__(256) void out_gemm(const unsigned short* __restrict__ A,
                                                const unsigned short* __restrict__ WtO,
                                                const float* __restrict__ opb,
                                                const float* __restrict__ qf,
                                                float* __restrict__ qpre) {
  int m0  = blockIdx.y * 16;           // N covered entirely (256)
  int tid = threadIdx.x, lane = tid & 31, wv = tid >> 5;
  __shared__ __align__(16) unsigned short As[16 * 256];   // 8 KB
  __shared__ __align__(16) float Csf[16 * 256];           // 16 KB fp32 staging
  f32x8 acc0 = {}, acc1 = {};

  const unsigned short* b0p = WtO + (size_t)(wv * 32) * TOTAL;
  const unsigned short* b1p = b0p + (size_t)16 * TOTAL;

  for (int kk = 0; kk < TOTAL; kk += 256) {
#pragma unroll
    for (int j = 0; j < 2; ++j) {
      int u  = tid + j * 256;
      int rr = u >> 5, c8 = (u & 31) * 8;
      int m  = m0 + rr;
      uint4 g = (m < BN) ? *(const uint4*)&A[(size_t)m * TOTAL + kk + c8] : uint4{0, 0, 0, 0};
      *(uint4*)&As[rr * 256 + c8] = g;
    }
    if (kk + 256 < TOTAL)
      __builtin_prefetch(&A[(size_t)(m0 + (tid >> 4)) * TOTAL + kk + 256 + ((tid & 15) << 4)], 0, 1);
    __syncthreads();
#pragma unroll
    for (int t = 0; t < 8; ++t) {
      bf16x16 a  = load_a_frag(As + t * 32, 256, lane);
      bf16x16 b0 = load_bT_frag(b0p + kk + t * 32, TOTAL, lane);
      bf16x16 b1 = load_bT_frag(b1p + kk + t * 32, TOTAL, lane);
      acc0 = wmma_bf16(a, b0, acc0);
      acc1 = wmma_bf16(a, b1, acc1);
    }
    __syncthreads();
  }
  // epilogue: stage fp32 tile, add bias + residual, coalesced float4 stores
#pragma unroll
  for (int rI = 0; rI < 8; ++rI) {
    int ml = rI + 8 * (lane >> 4);
    Csf[ml * 256 + wv * 32 + (lane & 15)]      = acc0[rI];
    Csf[ml * 256 + wv * 32 + 16 + (lane & 15)] = acc1[rI];
  }
  __syncthreads();
#pragma unroll
  for (int j = 0; j < 4; ++j) {
    int u  = tid + j * 256;              // 1024 float4 groups
    int rr = u >> 6, c4 = (u & 63) * 4;
    int m  = m0 + rr;
    if (m < BN) {
      float4 v = *(const float4*)&Csf[rr * 256 + c4];
      float4 b = *(const float4*)&opb[c4];
      float4 q = *(const float4*)&qf[(size_t)m * 256 + c4];
      float4 o;
      o.x = v.x + b.x + q.x; o.y = v.y + b.y + q.y;
      o.z = v.z + b.z + q.z; o.w = v.w + b.w + q.w;
      *(float4*)&qpre[(size_t)m * 256 + c4] = o;
    }
  }
}

// ---------------------------------------------------------------------------
// K7: final LayerNorm over C=256, scale/shift, write d_out (fp32)
// ---------------------------------------------------------------------------
__global__ __launch_bounds__(256) void ln_out(const float* __restrict__ qpre,
                                              const float* __restrict__ lng,
                                              const float* __restrict__ lnb,
                                              float* __restrict__ out) {
  int bn = blockIdx.x, t = threadIdx.x, lane = t & 31, wv = t >> 5;
  __shared__ float sred[16];
  __shared__ float sstat[2];
  float v = qpre[(size_t)bn * CQ + t];
  float ls = wave_sum(v);
  float lq = wave_sum(v * v);
  if (lane == 0) { sred[wv] = ls; sred[8 + wv] = lq; }
  __syncthreads();
  if (t == 0) {
    float s = 0.f, q = 0.f;
    for (int i = 0; i < 8; ++i) { s += sred[i]; q += sred[8 + i]; }
    float mean = s * (1.f / 256.f);
    float var  = q * (1.f / 256.f) - mean * mean;
    sstat[0] = mean; sstat[1] = rsqrtf(var + 1e-5f);
  }
  __syncthreads();
  out[(size_t)bn * CQ + t] = (v - sstat[0]) * sstat[1] * lng[t] + lnb[t];
}

// ---------------------------------------------------------------------------
extern "C" void kernel_launch(void* const* d_in, const int* in_sizes, int n_in,
                              void* d_out, int out_size, void* d_ws, size_t ws_size,
                              hipStream_t stream) {
  (void)in_sizes; (void)n_in; (void)out_size; (void)ws_size;
  const float* x0   = (const float*)d_in[0];
  const float* x1   = (const float*)d_in[1];
  const float* x2   = (const float*)d_in[2];
  const float* x3   = (const float*)d_in[3];
  const float* qf   = (const float*)d_in[4];
  const float* roi  = (const float*)d_in[5];
  const float* woff = (const float*)d_in[6];
  const float* boff = (const float*)d_in[7];
  const float* pgw  = (const float*)d_in[8];
  const float* pgb  = (const float*)d_in[9];
  const float* opw  = (const float*)d_in[10];
  const float* opb  = (const float*)d_in[11];
  const float* lng  = (const float*)d_in[12];
  const float* lnb  = (const float*)d_in[13];
  float* out = (float*)d_out;

  // workspace layout (256B aligned), total ~125 MB
  char* ws = (char*)d_ws;
  size_t off = 0;
  auto take = [&](size_t bytes) -> char* {
    char* p = ws + off;
    off = (off + bytes + 255) & ~(size_t)255;
    return p;
  };
  unsigned short* qf_bf   = (unsigned short*)take((size_t)BN * CQ * 2);
  unsigned short* pgw_t   = (unsigned short*)take((size_t)CQ * TOTAL * 2);  // (32768, 256)
  unsigned short* opw_t   = (unsigned short*)take((size_t)TOTAL * CQ * 2);  // (256, 32768)
  float*          coords  = (float*)take((size_t)BN * 128 * 6 * 4);
  unsigned short* sampled = (unsigned short*)take((size_t)BN * GG * PIN * CGc * 2);
  unsigned short* params  = (unsigned short*)take((size_t)BN * TOTAL * 2);
  unsigned short* hflat   = (unsigned short*)take((size_t)BN * TOTAL * 2);
  float*          qpre    = (float*)take((size_t)BN * CQ * 4);

  // K1: bf16 conversions; weights transposed to N-major for the GEMM B path
  cvt_bf16<<<160, 256, 0, stream>>>(qf, qf_bf, BN * CQ / 4);
  transpose_cvt<<<dim3(TOTAL / 64, CQ / 64), 256, 0, stream>>>(pgw, pgw_t, CQ, TOTAL);
  transpose_cvt<<<dim3(CQ / 64, TOTAL / 64), 256, 0, stream>>>(opw, opw_t, TOTAL, CQ);

  // K2: offsets + coords + level weights
  offset_kernel<<<BN, 128, 0, stream>>>(qf, roi, woff, boff, coords);

  // K3: params GEMM (WMMA bf16, barrier-free inner loop)
  param_gemm<<<dim3(TOTAL / 256, (BN + 15) / 16), 256, 0, stream>>>(qf_bf, pgw_t, pgb, params);

  // K4: bilinear sampling
  sample_kernel<<<dim3(BN, GG), 256, 0, stream>>>(x0, x1, x2, x3, coords, sampled);

  // K5: per-query mixing (WMMA bf16 x2 + LN/ReLU)
  mix_kernel<<<dim3(BN, GG), 256, 0, stream>>>(sampled, params, hflat);

  // K6: output GEMM + residual (WMMA bf16)
  out_gemm<<<dim3(1, (BN + 15) / 16), 256, 0, stream>>>(hflat, opw_t, opb, qf, qpre);

  // K7: final LayerNorm
  ln_out<<<BN, 256, 0, stream>>>(qpre, lng, lnb, out);
}
